// GIN_72756745994408
// MI455X (gfx1250) — matrix operand order repaired and checked by reading, compiled, not run on hardware
//
#include <hip/hip_runtime.h>
#include <hip/hip_bf16.h>

typedef __attribute__((ext_vector_type(16))) _Float16 v16h;
typedef __attribute__((ext_vector_type(8)))  float    v8f;
typedef __attribute__((ext_vector_type(4)))  unsigned int u32x4;
typedef __attribute__((ext_vector_type(8)))  int      i32x8;
typedef __attribute__((ext_vector_type(4)))  int      i32x4;

#define B_GRAPHS 4096
#define NODES    32
#define VOCAB    100
#define EMB      64
#define GDIM     64
#define E_PER    512
#define NTOT     (B_GRAPHS * NODES)
#define NEDGE    (B_GRAPHS * E_PER)
#define CDIM     192   /* 3 * GDIM */
#define HID      384   /* CDIM * 2 */
#define LN_EPS   1e-5f

#if defined(__has_builtin)
#if __has_builtin(__builtin_amdgcn_tensor_load_to_lds) && __has_builtin(__builtin_amdgcn_s_wait_tensorcnt)
#define HAVE_TDM 1
#endif
#endif
#ifndef HAVE_TDM
#define HAVE_TDM 0
#endif

// ---------------------------------------------------------------------------
// Re-tile an f32 weight matrix [Ktot x Ntot] (row-major, K = in-features) into
// native WMMA B-fragment layout for v_wmma_f32_16x16x32_f16:
//   dst[((kt*NT + nt)*32 + lane)*16 + j] = W[kt*32 + 8*(lane>>4) + (j&7) + 16*(j>>3)][nt*16 + (lane&15)]
// ---------------------------------------------------------------------------
__global__ void prep_tiles(const float* __restrict__ W, _Float16* __restrict__ dst,
                           int Ktot, int Ntot) {
    int NT = Ntot >> 4;
    int total = (Ktot >> 5) * NT * 32 * 16;
    for (int i = blockIdx.x * blockDim.x + threadIdx.x; i < total;
         i += gridDim.x * blockDim.x) {
        int j    = i & 15;
        int l    = (i >> 4) & 31;
        int tile = i >> 9;
        int nt   = tile % NT;
        int kt   = tile / NT;
        int K    = kt * 32 + ((l >> 4) << 3) + (j & 7) + ((j >> 3) << 4);
        int Ncol = nt * 16 + (l & 15);
        dst[i] = (_Float16)W[K * Ntot + Ncol];
    }
}

// ---------------------------------------------------------------------------
// Embedding gather: h[b*32+n][f] = emb_tables[n][xf[b,n]][f]
// ---------------------------------------------------------------------------
__global__ void embed_kernel(const int* __restrict__ x, const int* __restrict__ los,
                             const float* __restrict__ emb, float* __restrict__ h) {
    int b = blockIdx.x;
    for (int i = threadIdx.x; i < NODES * EMB; i += blockDim.x) {
        int n = i >> 6, f = i & 63;
        int v = (n < NODES - 1) ? x[b * (NODES - 1) + n] : los[b];
        h[((size_t)b * NODES + n) * EMB + f] = emb[((size_t)n * VOCAB + v) * EMB + f];
    }
}

// LDS aperture addresses keep the byte offset in the low 32 bits (ISA 10.2).
__device__ __forceinline__ unsigned lds_off(const void* p) {
    return (unsigned)(unsigned long long)p;
}

#if HAVE_TDM
// Issue one TDM descriptor: copy a [rows x width] f32/int32 tile (row stride
// `stride` elements) from global memory into LDS at byte offset `lds_addr`.
__device__ __forceinline__ void tdm_load_2d(unsigned lds_addr, const void* gptr,
                                            unsigned width, unsigned rows,
                                            unsigned stride) {
    unsigned long long ga = (unsigned long long)gptr;
    u32x4 g0;
    g0[0] = 1u;                                   // count=1, user descriptor
    g0[1] = lds_addr;                             // lds_addr[31:0]
    g0[2] = (unsigned)(ga & 0xFFFFFFFFu);         // global_addr[31:0]
    g0[3] = (unsigned)(ga >> 32) | (2u << 30);    // global_addr[56:32] | type=2
    i32x8 g1;
    g1[0] = (int)(2u << 16);                      // data_size = 4 bytes
    g1[1] = (int)((width & 0xFFFFu) << 16);       // tensor_dim0[15:0]
    g1[2] = (int)((width >> 16) | ((rows & 0xFFFFu) << 16)); // dim0 hi | dim1 lo
    g1[3] = (int)((rows >> 16) | ((width & 0xFFFFu) << 16)); // dim1 hi | tile_dim0
    g1[4] = (int)(rows > 1 ? rows : 0);           // tile_dim1 (0 = unused)
    g1[5] = (int)stride;                          // tensor_dim0_stride[31:0]
    g1[6] = 0;                                    // stride0 hi | stride1 lo
    g1[7] = 0;                                    // stride1 hi
    i32x4 gz4 = {0, 0, 0, 0};
    i32x8 gz8 = {0, 0, 0, 0, 0, 0, 0, 0};
    __builtin_amdgcn_tensor_load_to_lds(g0, g1, gz4, gz4, gz8, 0);
}
#endif

// ---------------------------------------------------------------------------
// Fused GIN layer, one block per graph (128 threads = 4 wave32):
//   TDM tile DMA -> LDS scatter-add -> (1+eps)h+agg -> f16 -> WMMA GEMM1 ->
//   LN -> ReLU -> f16 -> WMMA GEMM2 -> h_out + pooled segment.
// ---------------------------------------------------------------------------
__global__ __launch_bounds__(128)
void gin_layer(const float* __restrict__ h_in, float* __restrict__ h_out,
               const int* __restrict__ e_src, const int* __restrict__ e_dst,
               const _Float16* __restrict__ W1t, const float* __restrict__ b1,
               const float* __restrict__ gam,  const float* __restrict__ bet,
               const _Float16* __restrict__ W2t, const float* __restrict__ b2,
               const float* __restrict__ eps_arr, int eps_idx,
               float* __restrict__ pooled, int pool_off) {
    __shared__ float    hL [NODES][GDIM];
    __shared__ float    agg[NODES][GDIM];
    __shared__ int      eL [E_PER * 2];
    __shared__ _Float16 z16[NODES][GDIM];
    __shared__ float    zg [NODES][GDIM];
    __shared__ float    mu_s[NODES], rs_s[NODES];

    const int b    = blockIdx.x;
    const int tid  = threadIdx.x;
    const int lane = tid & 31;
    const int wave = tid >> 5;
    const int bOff = b * NODES;

    const float* hbase = h_in + (size_t)b * NODES * GDIM;

#if HAVE_TDM
    // Wave 0 issues both DMA descriptors (TDM ignores EXEC; uniform branch
    // keeps waves 1..3 from issuing redundant copies), waits TENSORcnt==0,
    // then the workgroup barrier publishes the LDS tiles to all waves.
    if (wave == 0) {
        tdm_load_2d(lds_off(&hL[0][0]), hbase, NODES * GDIM, 1, NODES * GDIM);
        tdm_load_2d(lds_off(&eL[0]), e_src + b * E_PER, E_PER, 2, NEDGE);
        __builtin_amdgcn_s_wait_tensorcnt(0);
    }
    for (int i = tid; i < NODES * GDIM; i += 128)
        agg[i >> 6][i & 63] = 0.0f;
#else
    __builtin_prefetch(hbase, 0, 0);               // global_prefetch_b8
    for (int i = tid; i < NODES * GDIM; i += 128) {
        hL [i >> 6][i & 63] = hbase[i];
        agg[i >> 6][i & 63] = 0.0f;
    }
    for (int i = tid; i < E_PER; i += 128) {
        eL[i]         = e_src[b * E_PER + i];
        eL[E_PER + i] = e_dst[b * E_PER + i];
    }
#endif
    __syncthreads();

    // LDS-atomic scatter-add (segment_sum): ds_add_f32
    {
        int f = tid & 63;
        for (int e = (tid >> 6); e < E_PER; e += 2) {
            int s = eL[e] - bOff;
            int d = eL[E_PER + e] - bOff;
            atomicAdd(&agg[d][f], hL[s][f]);
        }
    }
    __syncthreads();

    const float epsv = 1.0f + eps_arr[eps_idx];
    for (int i = tid; i < NODES * GDIM; i += 128) {
        int r = i >> 6, f = i & 63;
        z16[r][f] = (_Float16)(epsv * hL[r][f] + agg[r][f]);
    }
    __syncthreads();

    // ---- GEMM1: z16[32x64] @ W1[64x64] -> zg (+b1) -------------------------
    for (int m = 0; m < 2; ++m) {
        v8f acc = {};
        for (int k = 0; k < 2; ++k) {
            v16h a, bm;
            int M  = m * 16 + (lane & 15);
            int kb = k * 32 + ((lane >> 4) << 3);
            #pragma unroll
            for (int j = 0; j < 16; ++j)
                a[j] = z16[M][kb + (j & 7) + ((j >> 3) << 4)];
            const _Float16* bt = W1t + (((k * 4 + wave) * 32 + lane) << 4);
            #pragma unroll
            for (int j = 0; j < 16; ++j) bm[j] = bt[j];
            acc = __builtin_amdgcn_wmma_f32_16x16x32_f16(false, a, false, bm,
                                                         (short)0, acc, false, false);
        }
        int Mb = m * 16 + ((lane >> 4) << 3);
        int Nc = wave * 16 + (lane & 15);
        #pragma unroll
        for (int r = 0; r < 8; ++r)
            zg[Mb + r][Nc] = acc[r] + b1[Nc];
    }
    __syncthreads();

    // ---- LayerNorm + ReLU -> z16 ------------------------------------------
    if (tid < NODES) {
        float s = 0.0f, s2 = 0.0f;
        for (int f = 0; f < GDIM; ++f) { float v = zg[tid][f]; s += v; s2 += v * v; }
        float mu  = s  * (1.0f / GDIM);
        float var = s2 * (1.0f / GDIM) - mu * mu;
        mu_s[tid] = mu;
        rs_s[tid] = rsqrtf(var + LN_EPS);
    }
    __syncthreads();
    for (int i = tid; i < NODES * GDIM; i += 128) {
        int r = i >> 6, f = i & 63;
        float v = (zg[r][f] - mu_s[r]) * rs_s[r] * gam[f] + bet[f];
        z16[r][f] = (_Float16)(v > 0.0f ? v : 0.0f);
    }
    __syncthreads();

    // ---- GEMM2: z16 @ W2 -> zg (+b2) --------------------------------------
    for (int m = 0; m < 2; ++m) {
        v8f acc = {};
        for (int k = 0; k < 2; ++k) {
            v16h a, bm;
            int M  = m * 16 + (lane & 15);
            int kb = k * 32 + ((lane >> 4) << 3);
            #pragma unroll
            for (int j = 0; j < 16; ++j)
                a[j] = z16[M][kb + (j & 7) + ((j >> 3) << 4)];
            const _Float16* bt = W2t + (((k * 4 + wave) * 32 + lane) << 4);
            #pragma unroll
            for (int j = 0; j < 16; ++j) bm[j] = bt[j];
            acc = __builtin_amdgcn_wmma_f32_16x16x32_f16(false, a, false, bm,
                                                         (short)0, acc, false, false);
        }
        int Mb = m * 16 + ((lane >> 4) << 3);
        int Nc = wave * 16 + (lane & 15);
        #pragma unroll
        for (int r = 0; r < 8; ++r)
            zg[Mb + r][Nc] = acc[r] + b2[Nc];
    }
    __syncthreads();

    // ---- write h_out + pooled node-sum ------------------------------------
    float* obase = h_out + (size_t)b * NODES * GDIM;
    for (int i = tid; i < NODES * GDIM; i += 128)
        obase[i] = zg[i >> 6][i & 63];
    if (tid < GDIM) {
        float s = 0.0f;
        for (int r = 0; r < NODES; ++r) s += zg[r][tid];
        pooled[(size_t)b * CDIM + pool_off + tid] = s;
    }
}

// ---------------------------------------------------------------------------
// Classifier: relu(pooled[4096x192] @ wc1[192x384] + bc1) @ wc2[384] + bc2
// One block = 16 rows; 8 waves x 3 N-tiles x 6 K-tiles = 144 WMMAs / block.
// ---------------------------------------------------------------------------
__global__ __launch_bounds__(256)
void classifier(const float* __restrict__ pooled, const _Float16* __restrict__ Wc1t,
                const float* __restrict__ bc1, const float* __restrict__ wc2,
                const float* __restrict__ bc2, float* __restrict__ out) {
    __shared__ _Float16 a16[16][CDIM];
    __shared__ float    zc [16][HID];
    __shared__ float    part[16][16];

    const int rb   = blockIdx.x * 16;
    const int tid  = threadIdx.x;
    const int lane = tid & 31;
    const int wave = tid >> 5;

    for (int i = tid; i < 16 * CDIM; i += 256)
        a16[i / CDIM][i % CDIM] = (_Float16)pooled[(size_t)(rb + i / CDIM) * CDIM + i % CDIM];
    __syncthreads();

    for (int t = 0; t < 3; ++t) {
        int nt = wave * 3 + t;
        v8f acc = {};
        for (int k = 0; k < 6; ++k) {
            v16h a, bm;
            int M  = lane & 15;
            int kb = k * 32 + ((lane >> 4) << 3);
            #pragma unroll
            for (int j = 0; j < 16; ++j)
                a[j] = a16[M][kb + (j & 7) + ((j >> 3) << 4)];
            const _Float16* bt = Wc1t + (((k * 24 + nt) * 32 + lane) << 4);
            #pragma unroll
            for (int j = 0; j < 16; ++j) bm[j] = bt[j];
            acc = __builtin_amdgcn_wmma_f32_16x16x32_f16(false, a, false, bm,
                                                         (short)0, acc, false, false);
        }
        int Mb = (lane >> 4) << 3;
        int Nc = nt * 16 + (lane & 15);
        #pragma unroll
        for (int r = 0; r < 8; ++r) {
            float v = acc[r] + bc1[Nc];
            zc[Mb + r][Nc] = v > 0.0f ? v : 0.0f;
        }
    }
    __syncthreads();

    {
        int row = tid >> 4, seg = tid & 15;
        float s = 0.0f;
        for (int f = seg * 24; f < seg * 24 + 24; ++f) s += zc[row][f] * wc2[f];
        part[row][seg] = s;
    }
    __syncthreads();
    if (tid < 16) {
        float s = 0.0f;
        for (int k = 0; k < 16; ++k) s += part[tid][k];
        out[rb + tid] = s + bc2[0];
    }
}

// ---------------------------------------------------------------------------
extern "C" void kernel_launch(void* const* d_in, const int* in_sizes, int n_in,
                              void* d_out, int out_size, void* d_ws, size_t ws_size,
                              hipStream_t stream) {
    (void)in_sizes; (void)n_in; (void)out_size; (void)ws_size;

    const int*   x     = (const int*)d_in[0];
    const int*   los   = (const int*)d_in[1];
    const int*   ei    = (const int*)d_in[2];
    const float* emb   = (const float*)d_in[3];
    const float* win1  = (const float*)d_in[4];
    const float* bin1  = (const float*)d_in[5];
    const float* gin_g = (const float*)d_in[6];
    const float* gin_b = (const float*)d_in[7];
    const float* win2  = (const float*)d_in[8];
    const float* bin2  = (const float*)d_in[9];
    const float* wh1   = (const float*)d_in[10];
    const float* bh1   = (const float*)d_in[11];
    const float* gh_g  = (const float*)d_in[12];
    const float* gh_b  = (const float*)d_in[13];
    const float* wh2   = (const float*)d_in[14];
    const float* bh2   = (const float*)d_in[15];
    const float* eps   = (const float*)d_in[16];
    const float* wc1   = (const float*)d_in[17];
    const float* bc1   = (const float*)d_in[18];
    const float* wc2   = (const float*)d_in[19];
    const float* bc2   = (const float*)d_in[20];
    float* outp = (float*)d_out;

    char* ws = (char*)d_ws;
    size_t off = 0;
    float* hA     = (float*)(ws + off); off += (size_t)NTOT * GDIM * 4;
    float* hB     = (float*)(ws + off); off += (size_t)NTOT * GDIM * 4;
    float* pooled = (float*)(ws + off); off += (size_t)B_GRAPHS * CDIM * 4;
    _Float16* w1t  = (_Float16*)(ws + off); off += 4096 * 2;   // 64x64 tiles
    _Float16* w2t  = (_Float16*)(ws + off); off += 4096 * 2;
    _Float16* wh1t = (_Float16*)(ws + off); off += 4096 * 2;
    _Float16* wh2t = (_Float16*)(ws + off); off += 4096 * 2;
    _Float16* wc1t = (_Float16*)(ws + off); off += 73728 * 2;  // 192x384 tiles

    prep_tiles<<<16,  256, 0, stream>>>(win1, w1t, 64, 64);
    prep_tiles<<<16,  256, 0, stream>>>(win2, w2t, 64, 64);
    prep_tiles<<<16,  256, 0, stream>>>(wh1,  wh1t, 64, 64);
    prep_tiles<<<16,  256, 0, stream>>>(wh2,  wh2t, 64, 64);
    prep_tiles<<<288, 256, 0, stream>>>(wc1,  wc1t, CDIM, HID);

    embed_kernel<<<B_GRAPHS, 256, 0, stream>>>(x, los, emb, hA);

    const int* e_src = ei;
    const int* e_dst = ei + NEDGE;
    gin_layer<<<B_GRAPHS, 128, 0, stream>>>(hA, hB, e_src, e_dst,
        w1t,  bin1, gin_g, gin_b, w2t,  bin2, eps, 0, pooled, 0);
    gin_layer<<<B_GRAPHS, 128, 0, stream>>>(hB, hA, e_src, e_dst,
        wh1t, bh1,  gh_g,  gh_b,  wh2t, bh2,  eps, 1, pooled, 64);
    gin_layer<<<B_GRAPHS, 128, 0, stream>>>(hA, hB, e_src, e_dst,
        wh1t, bh1,  gh_g,  gh_b,  wh2t, bh2,  eps, 2, pooled, 128);

    classifier<<<B_GRAPHS / 16, 256, 0, stream>>>(pooled, wc1t, bc1, wc2, bc2, outp);
}
